// HemiFusionDecoder_31018253812279
// MI455X (gfx1250) — matrix operand order repaired and checked by reading, not compiled, not run on hardware
//
#include <hip/hip_runtime.h>

// ---------------------------------------------------------------------------
// HemiFusionDecoder on MI455X (gfx1250, wave32, WMMA + TDM)
//
// * stride-2 ConvTranspose2d -> 4 output phases, each a dense implicit GEMM
//   (K = Cin*(k/2)^2) on v_wmma_f32_16x16x32_f16, f32 accumulate.
// * Weights pre-packed once into phase/tile-major f16 tiles (L2-resident);
//   A tiles streamed into LDS by the Tensor Data Mover (pad_enable gives the
//   conflict-free 80B row stride), double-buffered, one barrier per K-step.
// * B gather is branchless with all address math hoisted out of the K loop.
// * Cout=1 hemisphere layer: tap-expansion GEMM (M = 64 taps, all useful)
//   + gather epilogue.
// ---------------------------------------------------------------------------

typedef _Float16 f16;
typedef __attribute__((ext_vector_type(16))) _Float16 v16h;
typedef __attribute__((ext_vector_type(8)))  _Float16 v8h;
typedef __attribute__((ext_vector_type(8)))  float    v8f;
typedef __attribute__((ext_vector_type(4)))  unsigned int v4u;
typedef __attribute__((ext_vector_type(8)))  int v8i;
typedef __attribute__((ext_vector_type(4)))  int v4i;

union Frag { v16h v; v8h h[2]; };

#define LROW 40            // LDS row stride in halves: 32 payload + 8 pad (80B)
#define BUFH (64 * LROW)   // one LDS buffer (64 rows)

#if __has_builtin(__builtin_amdgcn_tensor_load_to_lds) && \
    __has_builtin(__builtin_amdgcn_s_wait_tensorcnt)
#define HAVE_TDM 1
#endif
#if __has_include(<hip/amd_detail/amd_gfx1250_TDM.h>)
#define TDM_ARITY6 1
#endif

#if HAVE_TDM
// Load one contiguous 4KB packed-weight tile (512 x 8B) into LDS, inserting
// 4 DWORDs of padding after every 16 DWORDs -> rows of 32 halves at stride 40.
__device__ __forceinline__ void tdm_load_a(unsigned ldsOff, const f16* gsrc)
{
  unsigned long long ga = (unsigned long long)(size_t)(const void*)gsrc;
  v4u g0 = { 1u,                                    // count=1, user mode
             ldsOff,                                // lds_addr (bytes)
             (unsigned)ga,                          // global_addr[31:0]
             (unsigned)((ga >> 32) & 0x1ffffffull)  // global_addr[56:32]
               | 0x80000000u };                     // type=2 ("image")
  v8i g1 = { (int)((3u << 16)   // data_size = 8B
               | (1u << 20)     // pad_enable
               | (3u << 22)     // pad_interval: 16 DWORDs
               | (3u << 25)),   // pad_amount:   4 DWORDs
             (int)(512u << 16), // tensor_dim0 = 512 (8B units), bits[79:48]
             (int)(1u << 16),   // tensor_dim1 = 1,              bits[111:80]
             (int)(512u << 16), // tile_dim0   = 512,            bits[127:112]
             1,                 // tile_dim1   = 1
             512,               // tensor_dim0_stride = 512
             0, 0 };
  v4i z4 = { 0, 0, 0, 0 };
#if TDM_ARITY6
  v8i z8 = { 0, 0, 0, 0, 0, 0, 0, 0 };
  __builtin_amdgcn_tensor_load_to_lds(g0, g1, z4, z4, z8, 0);
#else
  __builtin_amdgcn_tensor_load_to_lds(g0, g1, z4, z4, 0);
#endif
}
#endif

// ------------------------------ dense + act --------------------------------
__global__ __launch_bounds__(256) void dense_act_kernel(
    const float* __restrict__ z, const float* __restrict__ w,
    const float* __restrict__ b, f16* __restrict__ out,
    int B, int K, int F)
{
  int i = blockIdx.x * 256 + threadIdx.x;
  if (i >= B * F) return;
  int bb = i / F, f = i - bb * F;
  float acc = b[f];
  const float* zr = z + bb * K;
  for (int k = 0; k < K; ++k) acc += zr[k] * w[k * F + f];
  out[i] = (f16)(acc > 0.f ? acc : 0.01f * acc);
}

// --------------------- weight pre-pack (runs once per layer) ---------------
// apk[((phase*(Cout/64)+mt)*nKt+ktile)*2048 + m*32 + kkl] = f16 tap weight
__global__ __launch_bounds__(256) void pack_weights(
    const float* __restrict__ wgt, f16* __restrict__ apk,
    int Cout, int Ktot, int KK, int pad)
{
  int i = blockIdx.x * 256 + threadIdx.x;
  if (i >= 4 * Cout * Ktot) return;
  int tapsLog = (KK == 8) ? 2 : 1, tap2Log = 2 * tapsLog;
  int taps = 1 << tapsLog, tap2 = 1 << tap2Log;
  int nKt = Ktot >> 5;
  int kkl = i & 31, m = (i >> 5) & 63;
  int r = i >> 11;
  int ktile = r % nKt; r /= nKt;
  int mt = r % (Cout >> 6); int phase = r / (Cout >> 6);
  int py = (phase >> 1) & 1, px = phase & 1;
  int P = KK - 1 - pad;
  int parY = (P + py) & 1, parX = (P + px) & 1;
  int co = mt * 64 + m, kk = ktile * 32 + kkl;
  int ci = kk >> tap2Log, t = kk & (tap2 - 1);
  int ty = t >> tapsLog, tx = t & (taps - 1);
  int wrow = KK - 1 - (2 * ty + parY), wcol = KK - 1 - (2 * tx + parX);
  apk[i] = (f16)wgt[((ci * Cout + co) * KK + wrow) * KK + wcol];
}

// -------------------- phase-decomposed tconv, implicit GEMM ----------------
// Block tile 64(M=Cout) x 64(N=pixels); 8 waves, 2 WMMAs each; double-
// buffered LDS; A tile via TDM (or b128 copy fallback); one barrier/K-step.
__global__ __launch_bounds__(256) void tconv_phase_wmma(
    const f16* __restrict__ xin,   // (B, CinTot, H, W) post-act f16
    const f16* __restrict__ apk,   // packed weights (see pack_weights)
    const float* __restrict__ bias,
    void* __restrict__ yout,
    int Bn, int CinTot, int ci0, int Cin, int Cout,
    int H, int W, int KK, int pad, int zp,
    int Hout, int Wout, int finalF32)
{
  const int phase = blockIdx.z;
  const int py = (phase >> 1) & 1, px = phase & 1;
  const int tapsLog = (KK == 8) ? 2 : 1, tap2Log = 2 * tapsLog;
  const int taps = 1 << tapsLog, tap2 = 1 << tap2Log;
  const int P = KK - 1 - pad;
  const int parY = (P + py) & 1, parX = (P + px) & 1;
  const int oh2 = Hout >> 1, ow2 = Wout >> 1, hw2 = oh2 * ow2;
  const int Ntot = Bn * hw2, Ktot = Cin << tap2Log;
  const int nKt = Ktot >> 5, HW = H * W;
  const int mBase = blockIdx.y * 64, nBase = blockIdx.x * 64;

  __shared__ __align__(16) f16 As[2 * BUFH];
  __shared__ __align__(16) f16 Bs[2 * BUFH];

  const int tid = threadIdx.x, lane = tid & 31, wave = tid >> 5;
  const int wtm = wave >> 1, wtn = wave & 1;
  const int half = lane >> 4, l16 = lane & 15;

  // ---- per-thread B-gather bases, K-invariant (tap2 divides 32) ----
  const int nb = tid & 63;            // pixel within N tile
  const int kk0 = (tid >> 6) * 8;     // this thread's 8 consecutive kk
  int  fidx[8];
  bool fval[8];
  {
    int ng = nBase + nb;
    bool ngv = ng < Ntot;
    if (!ngv) ng = Ntot - 1;
    int b = ng / hw2, rem = ng - b * hw2;
    int om = rem / ow2, on = rem - om * ow2;
    int bofs = (b * CinTot + ci0) * HW;
    for (int j = 0; j < 8; ++j) {
      int kkl = kk0 + j;
      int t = kkl & (tap2 - 1);
      int ty = t >> tapsLog, tx = t & (taps - 1);
      int ky = 2 * ty + parY, kx = 2 * tx + parX;
      int riy = om + ((py + ky - P) >> 1) - zp;
      int rix = on + ((px + kx - P) >> 1) - zp;
      bool v = ngv && riy >= 0 && riy < H && rix >= 0 && rix < W;
      fval[j] = v;
      fidx[j] = v ? (bofs + (kkl >> tap2Log) * HW + riy * W + rix) : 0;
    }
  }

  const long aBase = ((long)((phase * (Cout >> 6) + blockIdx.y) * nKt)) << 11;

  auto stageA = [&](int kt, int pb) {
#if HAVE_TDM
    if (wave == 0)
      tdm_load_a((unsigned)(size_t)(const void*)&As[pb * BUFH],
                 apk + aBase + ((long)(kt >> 5) << 11));
#else
    const f16* src = apk + aBase + ((long)(kt >> 5) << 11) + tid * 8;
    int m = tid >> 2, c = tid & 3;
    *(v8h*)&As[pb * BUFH + m * LROW + c * 8] = *(const v8h*)src;
#endif
  };
  auto stageB = [&](int kt, int pb) {
    int ciStep = (kt >> tap2Log) * HW;
    union { v8h v; f16 e[8]; } pk;
    for (int j = 0; j < 8; ++j) {             // 8 branchless gathers
      int idx = fval[j] ? (fidx[j] + ciStep) : 0;
      f16 x = xin[idx];
      pk.e[j] = fval[j] ? x : (f16)0;
    }
    *(v8h*)&Bs[pb * BUFH + nb * LROW + kk0] = pk.v;  // one b128 ds store
  };

  v8f acc0 = {0.f,0.f,0.f,0.f,0.f,0.f,0.f,0.f};
  v8f acc1 = {0.f,0.f,0.f,0.f,0.f,0.f,0.f,0.f};

  int p = 0;
  stageA(0, 0);
  stageB(0, 0);
  for (int kt = 0; kt < Ktot; kt += 32) {
#if HAVE_TDM
    if (wave == 0) __builtin_amdgcn_s_wait_tensorcnt(0);
#endif
    __syncthreads();                          // publish buffer p
    const f16* Ap = &As[p * BUFH];
    const f16* Bp = &Bs[p * BUFH];
    Frag a, b0, b1;
    const int arow = wtm * 16 + l16;
    a.h[0]  = *(const v8h*)&Ap[arow * LROW + half * 8];
    a.h[1]  = *(const v8h*)&Ap[arow * LROW + 16 + half * 8];
    const int br0 = wtn * 32 + l16;
    b0.h[0] = *(const v8h*)&Bp[br0 * LROW + half * 8];
    b0.h[1] = *(const v8h*)&Bp[br0 * LROW + 16 + half * 8];
    b1.h[0] = *(const v8h*)&Bp[(br0 + 16) * LROW + half * 8];
    b1.h[1] = *(const v8h*)&Bp[(br0 + 16) * LROW + 16 + half * 8];
    int kn = kt + 32;
    if (kn < Ktot) {                          // prefetch next K-step
      stageA(kn, 1 - p);
      stageB(kn, 1 - p);
    }
    acc0 = __builtin_amdgcn_wmma_f32_16x16x32_f16(
        false, a.v, false, b0.v, (short)0, acc0, false, false);
    acc1 = __builtin_amdgcn_wmma_f32_16x16x32_f16(
        false, a.v, false, b1.v, (short)0, acc1, false, false);
    p ^= 1;
  }

  // ---- epilogue: bias (+ LeakyReLU for hidden layers), phase scatter ----
  for (int c = 0; c < 2; ++c) {
    const v8f& A = c ? acc1 : acc0;
    int n = nBase + wtn * 32 + c * 16 + l16;
    if (n < Ntot) {
      int b = n / hw2, rem = n - b * hw2;
      int om = rem / ow2, on = rem - om * ow2;
      int oy = 2 * om + py, ox = 2 * on + px;
      for (int v = 0; v < 8; ++v) {
        int co = mBase + wtm * 16 + half * 8 + v;  // D: M = v + 8*(lane>=16)
        if (co < Cout) {
          float r = A[v] + bias[co];
          int idx = ((b * Cout + co) * Hout + oy) * Wout + ox;
          if (finalF32) ((float*)yout)[idx] = r;
          else          ((f16*)yout)[idx] = (f16)(r > 0.f ? r : 0.01f * r);
        }
      }
    }
  }
}

// -------- hemisphere (Cout=1, k=8): tap-expansion GEMM, M = 64 taps --------
__global__ __launch_bounds__(256) void hemi_partial_wmma(
    const f16* __restrict__ xin,   // (8, 64, 244, 244) post-act f16
    const float* __restrict__ wgt, // (32, 1, 8, 8) f32
    f16* __restrict__ part,        // (8*244*244, 64)
    int ci0)
{
  __shared__ __align__(16) f16 As[BUFH];   // [tap][ci]
  __shared__ __align__(16) f16 Bs[BUFH];   // [n][ci] (32 rows used)
  const int tid = threadIdx.x, lane = tid & 31, wave = tid >> 5;
  const int wtm = wave >> 1, wtn = wave & 1;
  const int half = lane >> 4, l16 = lane & 15;
  const int nBase = blockIdx.x * 32;
  const int HW = 244 * 244;

  for (int j = 0; j < 8; ++j) {            // A: 64 taps x 32 ci
    int id = tid + 256 * j;
    int t = id >> 5, ci = id & 31;
    As[t * LROW + ci] = (f16)wgt[ci * 64 + t];
  }
  for (int j = 0; j < 4; ++j) {            // B: 32 pixels x 32 ci, lanes step ci (stride HW)
    int id = tid + 256 * j;
    int n = id >> 5, ci = id & 31;
    int ng = nBase + n;
    int b = ng / HW, rem = ng - b * HW;
    Bs[n * LROW + ci] = xin[(b * 64 + ci0 + ci) * HW + rem];
  }
  __syncthreads();

  Frag a, bf;
  const int arow = wtm * 16 + l16;
  a.h[0] = *(const v8h*)&As[arow * LROW + half * 8];
  a.h[1] = *(const v8h*)&As[arow * LROW + 16 + half * 8];
  const int brow = wtn * 16 + l16;
  bf.h[0] = *(const v8h*)&Bs[brow * LROW + half * 8];
  bf.h[1] = *(const v8h*)&Bs[brow * LROW + 16 + half * 8];
  v8f acc = {0.f,0.f,0.f,0.f,0.f,0.f,0.f,0.f};
  acc = __builtin_amdgcn_wmma_f32_16x16x32_f16(
      false, a.v, false, bf.v, (short)0, acc, false, false);

  int n = nBase + wtn * 16 + l16;
  for (int v = 0; v < 8; ++v) {
    int t = wtm * 16 + half * 8 + v;
    part[n * 64 + t] = (f16)acc[v];
  }
}

__global__ __launch_bounds__(256) void hemi_epilogue(
    const f16* __restrict__ part, const float* __restrict__ bias,
    float* __restrict__ out, int outOff)
{
  int i = blockIdx.x * 256 + threadIdx.x;
  if (i >= 8 * 524 * 524) return;
  int b = i / (524 * 524), rem = i - b * (524 * 524);
  int oy = rem / 524, ox = rem - oy * 524;
  int py = oy & 1, px = ox & 1;
  int om = oy >> 1, on = ox >> 1;
  int parY = (4 + py) & 1, parX = (4 + px) & 1;   // P = 4
  float acc = bias[0];
  for (int ty = 0; ty < 4; ++ty) {
    int ky = 2 * ty + parY;
    int riy = om + ((py + ky - 4) >> 1) - 9;      // zero_pad = 9
    if (riy < 0 || riy >= 244) continue;
    for (int tx = 0; tx < 4; ++tx) {
      int kx = 2 * tx + parX;
      int rix = on + ((px + kx - 4) >> 1) - 9;
      if (rix < 0 || rix >= 244) continue;
      int t = (7 - ky) * 8 + (7 - kx);
      acc += (float)part[((b * 244 + riy) * 244 + rix) * 64 + t];
    }
  }
  out[outOff + i] = acc;
}

// ------------------------------- launcher ----------------------------------
static void launch_layer(const f16* x, const float* w, f16* apk,
                         const float* b, void* y,
                         int Bn, int CinTot, int Cin, int Cout,
                         int H, int W, int KK, int pad, int zp,
                         int Hout, int Wout, int finalF32, hipStream_t s)
{
  int tap2 = (KK / 2) * (KK / 2);
  int Ktot = Cin * tap2;
  int nPk = 4 * Cout * Ktot;
  pack_weights<<<(nPk + 255) / 256, 256, 0, s>>>(w, apk, Cout, Ktot, KK, pad);
  int Nt = Bn * (Hout / 2) * (Wout / 2);
  dim3 g((Nt + 63) / 64, Cout / 64, 4);
  tconv_phase_wmma<<<g, 256, 0, s>>>(x, apk, b, y, Bn, CinTot, 0, Cin, Cout,
                                     H, W, KK, pad, zp, Hout, Wout, finalF32);
}

extern "C" void kernel_launch(void* const* d_in, const int* in_sizes, int n_in,
                              void* d_out, int out_size, void* d_ws, size_t ws_size,
                              hipStream_t stream) {
  (void)in_sizes; (void)n_in; (void)out_size; (void)ws_size;
  const float* z  = (const float*)d_in[0];
  const float* dw = (const float*)d_in[1];
  const float* db = (const float*)d_in[2];
  const float* w5 = (const float*)d_in[3];  const float* b5 = (const float*)d_in[4];
  const float* w4 = (const float*)d_in[5];  const float* b4 = (const float*)d_in[6];
  const float* w3 = (const float*)d_in[7];  const float* b3 = (const float*)d_in[8];
  const float* w2 = (const float*)d_in[9];  const float* b2 = (const float*)d_in[10];
  const float* lw1 = (const float*)d_in[11]; const float* lb1 = (const float*)d_in[12];
  const float* rw1 = (const float*)d_in[13]; const float* rb1 = (const float*)d_in[14];
  float* out = (float*)d_out;

  // Workspace layout (f16 units), ~126 MB total; the 64-tap partial buffer
  // aliases dead x0..x3.
  f16* ws = (f16*)d_ws;
  f16* x0   = ws;                 //     32768
  f16* x1   = ws + 32768;         //    819200
  f16* x2   = ws + 851968;        //   2768896
  f16* x3   = ws + 3620864;       //  13778944 (ends 17399808)
  f16* part = ws;                 //  30482432 (aliases x0..x3)
  f16* x4   = ws + 30482432;      //  30482432
  f16* ap5  = ws + 60964864;      //   1048576
  f16* ap4  = ws + 62013440;      //    524288
  f16* ap3  = ws + 62537728;      //    262144
  f16* ap2  = ws + 62799872;      //    131072 (ends 62930944)

  dense_act_kernel<<<(8 * 4096 + 255) / 256, 256, 0, stream>>>(
      z, dw, db, x0, 8, 128, 4096);

  launch_layer(x0, w5, ap5, b5, x1, 8, 256, 256, 256,   4,   4, 4, 1, 3,  20,  20, 0, stream);
  launch_layer(x1, w4, ap4, b4, x2, 8, 256, 256, 128,  20,  20, 4, 1, 3,  52,  52, 0, stream);
  launch_layer(x2, w3, ap3, b3, x3, 8, 128, 128, 128,  52,  52, 4, 1, 3, 116, 116, 0, stream);
  launch_layer(x3, w2, ap2, b2, x4, 8, 128, 128,  64, 116, 116, 4, 1, 3, 244, 244, 0, stream);

  const int nPix = 8 * 244 * 244;                 // 476288, /32 exact
  const int nOut = 8 * 524 * 524;                 // 2196608 per hemisphere
  hemi_partial_wmma<<<nPix / 32, 256, 0, stream>>>(x4, lw1, part, 0);
  hemi_epilogue<<<(nOut + 255) / 256, 256, 0, stream>>>(part, lb1, out, 0);
  hemi_partial_wmma<<<nPix / 32, 256, 0, stream>>>(x4, rw1, part, 32);
  hemi_epilogue<<<(nOut + 255) / 256, 256, 0, stream>>>(part, rb1, out, nOut);
}
